// TimeStretch_33311766347850
// MI455X (gfx1250) — matrix operand order repaired and checked by reading, not compiled, not run on hardware
//
#include <hip/hip_runtime.h>
#include <hip/hip_bf16.h>

typedef __attribute__((ext_vector_type(16))) _Float16 v16h;
typedef __attribute__((ext_vector_type(8)))  _Float16 v8h;
typedef __attribute__((ext_vector_type(8)))  float    v8f;

#define PI_F 3.14159265358979323846f

// ---- problem constants (match reference) ----
constexpr int BATCH   = 64;
constexpr int BINS    = 512;
constexpr int FRAMES  = 64;      // inverse-STFT frames
constexpr int W_      = 1022;    // window length
constexpr int HOP     = 511;
constexpr int SIG_LEN = 33215;   // W + HOP*(FRAMES-1)
constexpr int OUT_LEN = 34250;
constexpr int N_SEG   = 36;
constexpr int STEP    = 950;     // SEG - OV
constexpr int SEG     = 1000;
constexpr int OV      = 50;
constexpr int DSHIFT  = 50;      // per-segment right shift
constexpr int FR2     = 66;      // forward-STFT frames
constexpr int KDIM    = 1024;    // padded GEMM K / coefficient dim
constexpr int M1      = BATCH * FRAMES;  // 4096  (256 M-tiles)
constexpr int M2      = BATCH * FR2;     // 4224  (264 M-tiles)

// ---- workspace layout (bytes) ----
constexpr size_t OFF_BINV   = 0;                      // 1024x1024 f16: rows = t (N-dim), cols = coeff k (K-dim)
constexpr size_t OFF_BFWD   = (size_t)2  << 20;       // 1024x1024 f16: rows = out bin n (N-dim), cols = t (K-dim)
constexpr size_t OFF_A1     = (size_t)4  << 20;       // 4096x1024 f16
constexpr size_t OFF_FRAMES = (size_t)12 << 20;       // 4096x1024 f32 (16MB); later reused for out_ts (64x34250 f32)
constexpr size_t OFF_SIG    = (size_t)28 << 20;       // 64x34250 f32
constexpr size_t OFF_A2     = (size_t)37 << 20;       // 4224x1024 f16

__device__ __forceinline__ float hannf(int t) {
    return 0.5f * (1.0f - cosf(2.0f * PI_F * (float)t / (float)W_));
}

// Inverse-DFT basis with inverse window folded in.
// binv[t*1024 + k]: coefficient multiplying packed input [Re(0..511) | Im(0..511)].
__global__ void k_build_binv(_Float16* __restrict__ binv) {
    int idx = blockIdx.x * blockDim.x + threadIdx.x;
    if (idx >= 1024 * 1024) return;
    int t = idx >> 10, k = idx & 1023;
    float v = 0.0f;
    if (t < W_) {
        // inverse_stft_window_fn: win / tiled overlap-added squared win
        float wt = hannf(t);
        int h = t % HOP;
        float w0 = hannf(h), w1 = hannf(h + HOP);
        float invw = wt / (w0 * w0 + w1 * w1);
        if (k < 512) {
            int r = (k * t) % W_;                       // exact angle reduction
            float c = (k == 0 || k == 511) ? (1.0f / W_) : (2.0f / W_);
            v = c * cosf(2.0f * PI_F * (float)r / (float)W_) * invw;
        } else {
            int kk = k - 512;
            if (kk != 0 && kk != 511) {                 // imag of DC/Nyquist ignored by irfft
                int r = (kk * t) % W_;
                v = -(2.0f / W_) * sinf(2.0f * PI_F * (float)r / (float)W_) * invw;
            }
        }
    }
    binv[idx] = (_Float16)v;
}

// Forward-DFT basis with Hann window folded in.
// bfwd[n*1024 + t]: n<512 -> Re bin n; n>=512 -> Im bin (n-512). K-dim is t (padded).
__global__ void k_build_bfwd(_Float16* __restrict__ bfwd) {
    int idx = blockIdx.x * blockDim.x + threadIdx.x;
    if (idx >= 1024 * 1024) return;
    int n = idx >> 10, t = idx & 1023;
    float v = 0.0f;
    if (t < W_) {
        float wt = hannf(t);
        if (n < 512) {
            int r = (n * t) % W_;
            v = wt * cosf(2.0f * PI_F * (float)r / (float)W_);
        } else {
            int nn = n - 512;
            int r = (nn * t) % W_;
            v = -wt * sinf(2.0f * PI_F * (float)r / (float)W_);
        }
    }
    bfwd[idx] = (_Float16)v;
}

// Pack x[B,512,64,2] -> A1[b*64+f][c*512+k] in f16.
__global__ void k_pack_a1(const float* __restrict__ x, _Float16* __restrict__ a1) {
    int idx = blockIdx.x * blockDim.x + threadIdx.x;
    if (idx >= M1 * KDIM) return;
    int row = idx >> 10, col = idx & 1023;
    int b = row >> 6, f = row & 63;
    int c = col >> 9, k = col & 511;
    a1[idx] = (_Float16)x[(((size_t)(b * 512 + k)) * 64 + f) * 2 + c];
}

// 2x2 register-blocked WMMA core: one wave computes a 32x32 output block
// (4 x 16x16 tiles). Each K-step loads 2 A + 2 B fragments feeding 4 WMMAs
// -> 1 fragment (2 b128) per WMMA. K loop unrolled x2 so the scheduler can
// hoist the next slab's loads above this slab's WMMAs (in-wave pipelining).
// A is row-major [*, K]; BT is [N, K] (each output column's K-run contiguous).
__device__ __forceinline__ void wmma_tile_2x2(const _Float16* __restrict__ A,
                                              const _Float16* __restrict__ BT,
                                              int mtile0, int ntile0, int K,
                                              v8f acc[2][2]) {
    const int lane = threadIdx.x & 31;
    const int r16  = lane & 15;   // A row / B column within tile
    const int hi   = lane >> 4;   // half-lane select per ISA fragment layout
    const _Float16* arow[2];
    const _Float16* brow[2];
#pragma unroll
    for (int i = 0; i < 2; ++i) {
        arow[i] = A  + (size_t)((mtile0 + i) * 16 + r16) * (size_t)K;
        brow[i] = BT + (size_t)((ntile0 + i) * 16 + r16) * (size_t)K;
    }
#pragma unroll
    for (int mi = 0; mi < 2; ++mi)
#pragma unroll
        for (int ni = 0; ni < 2; ++ni)
            acc[mi][ni] = (v8f){};

#pragma unroll 2
    for (int kk = 0; kk < K; kk += 32) {
        v16h a[2], b[2];
#pragma unroll
        for (int i = 0; i < 2; ++i) {
            // A 16x32 fragment: lane holds K = kk + hi*8 + [0..7] and kk + 16 + hi*8 + [0..7]
            v8h a0 = *(const v8h*)(arow[i] + kk + hi * 8);
            v8h a1 = *(const v8h*)(arow[i] + kk + 16 + hi * 8);
            // B 32x16 fragment: lane holds K = kk + hi*8 + [0..7] and kk + 16 + hi*8 + [0..7] for its column
            v8h b0 = *(const v8h*)(brow[i] + kk + hi * 8);
            v8h b1 = *(const v8h*)(brow[i] + kk + 16 + hi * 8);
#pragma unroll
            for (int j = 0; j < 8; ++j) {
                a[i][j] = a0[j]; a[i][8 + j] = a1[j];
                b[i][j] = b0[j]; b[i][8 + j] = b1[j];
            }
            // prefetch the slab after next (emits global_prefetch_b8)
            __builtin_prefetch(arow[i] + kk + 64, 0, 1);
            __builtin_prefetch(brow[i] + kk + 64, 0, 1);
        }
#pragma unroll
        for (int mi = 0; mi < 2; ++mi)
#pragma unroll
            for (int ni = 0; ni < 2; ++ni)
                acc[mi][ni] = __builtin_amdgcn_wmma_f32_16x16x32_f16(
                    /*neg_a=*/false, a[mi], /*neg_b=*/false, b[ni],
                    /*c_mod=*/(short)0, acc[mi][ni],
                    /*reuse_a=*/false, /*reuse_b=*/false);
    }
}

// GEMM1: frames[4096][1024] = A1[4096][1024] x BINV (cols t, zero-padded t>=1022).
// Block = 8 waves, each wave a 32x32 block: block covers 16 M-tiles x 2 N-tiles.
__global__ void k_gemm1(const _Float16* __restrict__ a1,
                        const _Float16* __restrict__ binv,
                        float* __restrict__ frames) {
    int wave   = threadIdx.x >> 5;
    int mtile0 = (blockIdx.y * 8 + wave) * 2;   // 256 M-tiles total
    int ntile0 = blockIdx.x * 2;                // 64 N-tiles total
    v8f acc[2][2];
    wmma_tile_2x2(a1, binv, mtile0, ntile0, KDIM, acc);
    int lane = threadIdx.x & 31, col = lane & 15, hi = lane >> 4;
#pragma unroll
    for (int mi = 0; mi < 2; ++mi)
#pragma unroll
        for (int ni = 0; ni < 2; ++ni)
#pragma unroll
            for (int v = 0; v < 8; ++v) {       // C layout: VGPR v -> M = v + 8*hi
                int m = (mtile0 + mi) * 16 + v + 8 * hi;
                frames[(size_t)m * KDIM + (ntile0 + ni) * 16 + col] = acc[mi][ni][v];
            }
}

// Overlap-add (HOP = W/2 -> exactly <=2 contributors) + zero-pad to OUT_LEN.
__global__ void k_overlap_add(const float* __restrict__ frames, float* __restrict__ sig) {
    int idx = blockIdx.x * blockDim.x + threadIdx.x;
    if (idx >= BATCH * OUT_LEN) return;
    int b = idx / OUT_LEN, s = idx - b * OUT_LEN;
    float v = 0.0f;
    if (s < SIG_LEN) {
        int f1 = s / HOP;
#pragma unroll
        for (int d = 0; d < 2; ++d) {
            int f = f1 - d;
            if (f >= 0 && f < FRAMES) {
                int o = s - f * HOP;
                if (o < W_) v += frames[((size_t)(b * FRAMES + f) << 10) + o];
            }
        }
    }
    sig[idx] = v;
}

// Crossfaded time-stretch: each t covered by <=2 segments; masks computed analytically.
__global__ void k_timestretch(const float* __restrict__ sig, float* __restrict__ out_ts) {
    int idx = blockIdx.x * blockDim.x + threadIdx.x;
    if (idx >= BATCH * OUT_LEN) return;
    int b = idx / OUT_LEN, t = idx - b * OUT_LEN;
    float v = 0.0f;
    int c0 = t / STEP;
#pragma unroll
    for (int d = 0; d < 2; ++d) {
        int n = c0 - d;
        if (n >= 0 && n < N_SEG) {
            int o = t - n * STEP;
            if (o >= 0 && o < SEG) {
                float m;
                if (o < OV)        m = (n == 0)         ? 1.0f : (float)(o + 1) * (1.0f / OV);
                else if (o < STEP) m = 1.0f;
                else               m = (n == N_SEG - 1) ? 1.0f : (float)(SEG - o) * (1.0f / OV);
                v += m * sig[(size_t)b * OUT_LEN + (t - n * DSHIFT)];
            }
        }
    }
    out_ts[idx] = v;
}

// Frame + pack for forward STFT: A2[b*66+f][t] = out_ts[b, f*HOP + t], pad t>=1022.
__global__ void k_pack_a2(const float* __restrict__ out_ts, _Float16* __restrict__ a2) {
    int idx = blockIdx.x * blockDim.x + threadIdx.x;
    if (idx >= M2 * KDIM) return;
    int row = idx >> 10, t = idx & 1023;
    int b = row / FR2, f = row - b * FR2;
    _Float16 v = (_Float16)0.0f;
    if (t < W_) v = (_Float16)out_ts[(size_t)b * OUT_LEN + f * HOP + t];
    a2[idx] = v;
}

// GEMM2: S[4224][1024] = A2 x BFWD, scattered straight into d_out[B,512,66,2].
// 264 M-tiles: last block's upper waves take a wave-uniform early exit.
__global__ void k_gemm2(const _Float16* __restrict__ a2,
                        const _Float16* __restrict__ bfwd,
                        float* __restrict__ out) {
    int wave   = threadIdx.x >> 5;
    int mtile0 = (blockIdx.y * 8 + wave) * 2;
    int ntile0 = blockIdx.x * 2;
    if (mtile0 >= M2 / 16) return;              // wave-uniform; EXEC all-1 for WMMA
    v8f acc[2][2];
    wmma_tile_2x2(a2, bfwd, mtile0, ntile0, KDIM, acc);
    int lane = threadIdx.x & 31, col = lane & 15, hi = lane >> 4;
#pragma unroll
    for (int mi = 0; mi < 2; ++mi)
#pragma unroll
        for (int ni = 0; ni < 2; ++ni)
#pragma unroll
            for (int v = 0; v < 8; ++v) {
                int m = (mtile0 + mi) * 16 + v + 8 * hi;  // < 4224
                int b = m / FR2, f = m - b * FR2;
                int n = (ntile0 + ni) * 16 + col;
                int c = n >> 9, k = n & 511;              // n<512 -> Re, else Im
                out[(((size_t)b * 512 + k) * FR2 + f) * 2 + c] = acc[mi][ni][v];
            }
}

extern "C" void kernel_launch(void* const* d_in, const int* in_sizes, int n_in,
                              void* d_out, int out_size, void* d_ws, size_t ws_size,
                              hipStream_t stream) {
    const float* x = (const float*)d_in[0];
    float* out = (float*)d_out;
    char* ws = (char*)d_ws;

    _Float16* binv   = (_Float16*)(ws + OFF_BINV);
    _Float16* bfwd   = (_Float16*)(ws + OFF_BFWD);
    _Float16* a1     = (_Float16*)(ws + OFF_A1);
    float*    frames = (float*)   (ws + OFF_FRAMES);
    float*    sig    = (float*)   (ws + OFF_SIG);
    float*    out_ts = (float*)   (ws + OFF_FRAMES);   // reuse: frames dead after overlap-add
    _Float16* a2     = (_Float16*)(ws + OFF_A2);

    const int TPB = 256;
    k_build_binv <<<(1024 * 1024) / TPB, TPB, 0, stream>>>(binv);
    k_build_bfwd <<<(1024 * 1024) / TPB, TPB, 0, stream>>>(bfwd);
    k_pack_a1    <<<(M1 * KDIM) / TPB, TPB, 0, stream>>>(x, a1);
    // GEMM1: 256 M-tiles / (8 waves * 2) = 16 blocks.y; 64 N-tiles / 2 = 32 blocks.x
    k_gemm1      <<<dim3(32, 16), TPB, 0, stream>>>(a1, binv, frames);
    k_overlap_add<<<(BATCH * OUT_LEN + TPB - 1) / TPB, TPB, 0, stream>>>(frames, sig);
    k_timestretch<<<(BATCH * OUT_LEN + TPB - 1) / TPB, TPB, 0, stream>>>(sig, out_ts);
    k_pack_a2    <<<(M2 * KDIM) / TPB, TPB, 0, stream>>>(out_ts, a2);
    // GEMM2: 264 M-tiles -> 132 wave slots -> ceil(132/8)=17 blocks.y (guarded)
    k_gemm2      <<<dim3(32, 17), TPB, 0, stream>>>(a2, bfwd, out);
}
